// Net_1245540516470
// MI455X (gfx1250) — matrix operand order repaired and checked by reading, not compiled, not run on hardware
//
#include <hip/hip_runtime.h>

#define NNODES 100000
#define NEDGES 1600000
#define IN_FEATS 16
#define H_FEAT 64
#define N_HEADS 3
#define N_LAYERS 8
#define NUM_CLASSES 6
#define HID 192            // H_FEAT * N_HEADS
#define NEG_SLOPE 0.01f
#define K1PAD 32           // layer-1 K padded 16 -> 32

typedef __attribute__((ext_vector_type(16))) __bf16 v16bf;
typedef __attribute__((ext_vector_type(8)))  float  v8f;

// ---------- helpers ----------
__device__ __forceinline__ unsigned short f2bf(float f) {
    unsigned u = __float_as_uint(f);
    unsigned r = u + 0x7FFFu + ((u >> 16) & 1u);   // round-to-nearest-even
    return (unsigned short)(r >> 16);
}
// order-preserving encode of float into unsigned for atomicMax
__device__ __forceinline__ unsigned fenc(float f) {
    unsigned u = __float_as_uint(f);
    return (u & 0x80000000u) ? ~u : (u | 0x80000000u);
}
__device__ __forceinline__ float fdec(unsigned u) {
    unsigned v = (u & 0x80000000u) ? (u & 0x7FFFFFFFu) : ~u;
    return __uint_as_float(v);
}
#define MENC_NEGINF 0x007FFFFFu   // fenc(-inf)

// ---------- weight / input prep ----------
__global__ void pack_input_kernel(const float* __restrict__ feat,
                                  unsigned short* __restrict__ out) {
    int tid = blockIdx.x * blockDim.x + threadIdx.x;
    if (tid >= NNODES * K1PAD) return;
    int n = tid >> 5, k = tid & 31;
    out[tid] = (k < IN_FEATS) ? f2bf(feat[n * IN_FEATS + k]) : (unsigned short)0;
}
__global__ void prep_w1_kernel(const float* __restrict__ W1,
                               unsigned short* __restrict__ out) {
    int tid = blockIdx.x * blockDim.x + threadIdx.x;   // [192][32]
    if (tid >= HID * K1PAD) return;
    int row = tid >> 5, k = tid & 31;                  // row = head*64+o ; W1 flat [3,64,16]
    out[tid] = (k < IN_FEATS) ? f2bf(W1[row * IN_FEATS + k]) : (unsigned short)0;
}
__global__ void prep_w_kernel(const float* __restrict__ W,
                              unsigned short* __restrict__ out, int total) {
    int tid = blockIdx.x * blockDim.x + threadIdx.x;   // W flat [7][192][192] == desired layout
    if (tid < total) out[tid] = f2bf(W[tid]);
}

// ---------- bf16 WMMA GEMM:  D[M x 192] = A[M x K] * B^T ----------
// Register-blocked: each wave computes a 16x64 strip (4 accumulators).
// A fragment loaded once per k-chunk, shared across 4 independent WMMA chains.
#define NT_PER_WAVE 4
__global__ void __launch_bounds__(256)
gemm_bf16_kernel(const unsigned short* __restrict__ Abf,  // [M,K] bf16 row-major
                 const unsigned short* __restrict__ Bbf,  // [192,K] bf16 row-major (weight rows)
                 float* __restrict__ D, int M, int K) {
    const int lane  = threadIdx.x & 31;
    const int wave  = threadIdx.x >> 5;
    const int mtile = blockIdx.x * 8 + wave;            // uniform per wave -> EXEC stays full
    if (mtile * 16 >= M) return;
    const int ntile0 = blockIdx.y * NT_PER_WAVE;        // 0,4,8
    const int half   = lane >> 4;
    const int l16    = lane & 15;

    const unsigned* Arow = reinterpret_cast<const unsigned*>(
        Abf + (size_t)(mtile * 16 + l16) * K);
    const unsigned* Brow[NT_PER_WAVE];
#pragma unroll
    for (int t = 0; t < NT_PER_WAVE; ++t)
        Brow[t] = reinterpret_cast<const unsigned*>(
            Bbf + (size_t)((ntile0 + t) * 16 + l16) * K);

    union Frag { v16bf v; unsigned u[8]; };
    v8f acc[NT_PER_WAVE];
#pragma unroll
    for (int t = 0; t < NT_PER_WAVE; ++t) acc[t] = {};

    for (int kk = 0; kk < K; kk += 32) {
        Frag a;
#pragma unroll
        for (int j = 0; j < 8; ++j) {
            // A 16x32 bf16 layout: vgpr j<4: K = half*8 + 2j ; j>=4: +16
            int ka = kk + ((j >> 2) << 4) + half * 8 + ((j & 3) << 1);
            a.u[j] = Arow[ka >> 1];
        }
#pragma unroll
        for (int t = 0; t < NT_PER_WAVE; ++t) {
            Frag b;
#pragma unroll
            for (int j = 0; j < 8; ++j) {
                // B 32x16 bf16 layout: vgpr j<4: K = half*8 + 2j ; j>=4: +16
                int kb = kk + ((j >> 2) << 4) + half * 8 + ((j & 3) << 1);
                b.u[j] = Brow[t][kb >> 1];
            }
            acc[t] = __builtin_amdgcn_wmma_f32_16x16x32_bf16(
                false, a.v, false, b.v, (short)0, acc[t], false, false);
        }
    }
    // D layout: vgpr r -> row = half*8 + r, col = l16
#pragma unroll
    for (int t = 0; t < NT_PER_WAVE; ++t) {
        float* Dt = D + (size_t)(mtile * 16) * HID + (ntile0 + t) * 16 + l16;
#pragma unroll
        for (int r = 0; r < 8; ++r)
            Dt[(size_t)(half * 8 + r) * HID] = acc[t][r];
    }
}

// ---------- attention logits per node ----------
__global__ void es_ed_kernel(const float* __restrict__ z, const float* __restrict__ al,
                             float* __restrict__ es, float* __restrict__ ed) {
    int tid = blockIdx.x * blockDim.x + threadIdx.x;   // tid = node*3 + h
    if (tid >= NNODES * N_HEADS) return;
    int h = tid % N_HEADS, node = tid / N_HEADS;
    const float* zr = z + (size_t)node * HID + h * H_FEAT;
    const float* as = al + h * 2 * H_FEAT;
    const float* ad = as + H_FEAT;
    float s = 0.f, d = 0.f;
#pragma unroll 8
    for (int o = 0; o < H_FEAT; ++o) { float zv = zr[o]; s += zv * as[o]; d += zv * ad[o]; }
    es[tid] = s; ed[tid] = d;
}

__global__ void init_m_den_kernel(unsigned* __restrict__ menc, float* __restrict__ den) {
    int tid = blockIdx.x * blockDim.x + threadIdx.x;
    if (tid >= NNODES * N_HEADS) return;
    menc[tid] = MENC_NEGINF;
    den[tid]  = 0.f;
}

__global__ void edge_logits_kernel(const int* __restrict__ src, const int* __restrict__ dst,
                                   const float* __restrict__ es, const float* __restrict__ ed,
                                   float* __restrict__ ebuf, unsigned* __restrict__ menc) {
    int tid = blockIdx.x * blockDim.x + threadIdx.x;   // tid = edge*3 + h
    if (tid >= NEDGES * N_HEADS) return;
    int h = tid % N_HEADS, e = tid / N_HEADS;
    int s = src[e], d = dst[e];
    float v = es[s * N_HEADS + h] + ed[d * N_HEADS + h];
    v = v > 0.f ? v : NEG_SLOPE * v;                   // leaky_relu
    ebuf[tid] = v;
    atomicMax(&menc[d * N_HEADS + h], fenc(v));
}

__global__ void finalize_m_kernel(unsigned* __restrict__ menc) {
    int tid = blockIdx.x * blockDim.x + threadIdx.x;
    if (tid >= NNODES * N_HEADS) return;
    float m = fdec(menc[tid]);
    if (!isfinite(m)) m = 0.f;                         // empty segments -> 0
    reinterpret_cast<float*>(menc)[tid] = m;           // reuse buffer as float
}

__global__ void edge_exp_kernel(const int* __restrict__ dst,
                                const float* __restrict__ ebuf, const float* __restrict__ m,
                                float* __restrict__ exbuf, float* __restrict__ den) {
    int tid = blockIdx.x * blockDim.x + threadIdx.x;
    if (tid >= NEDGES * N_HEADS) return;
    int h = tid % N_HEADS, e = tid / N_HEADS;
    int d = dst[e];
    float ex = expf(ebuf[tid] - m[d * N_HEADS + h]);
    exbuf[tid] = ex;
    atomicAdd(&den[d * N_HEADS + h], ex);
}

// one block (192 threads) per edge: coalesced z reads + coalesced hn atomics
__global__ void __launch_bounds__(HID)
edge_aggregate_kernel(const int* __restrict__ src, const int* __restrict__ dst,
                      const float* __restrict__ exbuf, const float* __restrict__ den,
                      const float* __restrict__ z, float* __restrict__ hn) {
    int e = blockIdx.x;
    int c = threadIdx.x;           // 0..191
    int h = c >> 6;
    int s = src[e], d = dst[e];
    float alpha = exbuf[e * N_HEADS + h] / fmaxf(den[d * N_HEADS + h], 1e-9f);
    atomicAdd(&hn[(size_t)d * HID + c], alpha * z[(size_t)s * HID + c]);
}

__global__ void elu_convert_kernel(const float* __restrict__ hn,
                                   float* __restrict__ hf, unsigned short* __restrict__ hbf) {
    int tid = blockIdx.x * blockDim.x + threadIdx.x;
    if (tid >= NNODES * HID) return;
    float v = hn[tid];
    float o = v > 0.f ? v : expm1f(v);                 // elu(alpha=1)
    hf[tid]  = o;
    hbf[tid] = f2bf(o);
}

__global__ void final_fc_kernel(const float* __restrict__ hf, const float* __restrict__ fcW,
                                const float* __restrict__ fcb, float* __restrict__ out) {
    int tid = blockIdx.x * blockDim.x + threadIdx.x;   // tid = n*6 + c
    if (tid >= NNODES * NUM_CLASSES) return;
    int c = tid % NUM_CLASSES, n = tid / NUM_CLASSES;
    const float* hr = hf + (size_t)n * HID;
    const float* wr = fcW + c * HID;
    float s = fcb[c];
#pragma unroll 8
    for (int k = 0; k < HID; ++k) s += hr[k] * wr[k];
    out[tid] = s;
}

// ---------- launcher ----------
extern "C" void kernel_launch(void* const* d_in, const int* in_sizes, int n_in,
                              void* d_out, int out_size, void* d_ws, size_t ws_size,
                              hipStream_t stream) {
    (void)in_sizes; (void)n_in; (void)out_size; (void)ws_size;
    const float* node_feat = (const float*)d_in[0];
    const int*   src       = (const int*)d_in[1];
    const int*   dst       = (const int*)d_in[2];
    const float* W1        = (const float*)d_in[3];
    const float* a1        = (const float*)d_in[4];
    const float* W         = (const float*)d_in[5];
    const float* a         = (const float*)d_in[6];
    const float* fcW       = (const float*)d_in[7];
    const float* fcb       = (const float*)d_in[8];

    // workspace carve (256B aligned)
    char* p = (char*)d_ws;
    auto carve = [&](size_t bytes) -> void* {
        void* r = (void*)p;
        p += (bytes + 255) & ~(size_t)255;
        return r;
    };
    float*          hf    = (float*)         carve((size_t)NNODES * HID * 4);
    unsigned short* hbf   = (unsigned short*)carve((size_t)NNODES * HID * 2);
    unsigned short* inpad = (unsigned short*)carve((size_t)NNODES * K1PAD * 2);
    float*          z     = (float*)         carve((size_t)NNODES * HID * 4);
    float*          hn    = (float*)         carve((size_t)NNODES * HID * 4);
    float*          es    = (float*)         carve((size_t)NNODES * N_HEADS * 4);
    float*          ed    = (float*)         carve((size_t)NNODES * N_HEADS * 4);
    unsigned*       menc  = (unsigned*)      carve((size_t)NNODES * N_HEADS * 4);
    float*          den   = (float*)         carve((size_t)NNODES * N_HEADS * 4);
    float*          ebuf  = (float*)         carve((size_t)NEDGES * N_HEADS * 4);
    float*          exbuf = (float*)         carve((size_t)NEDGES * N_HEADS * 4);
    unsigned short* w1bf  = (unsigned short*)carve((size_t)HID * K1PAD * 2);
    unsigned short* wbf   = (unsigned short*)carve((size_t)(N_LAYERS - 1) * HID * HID * 2);

    const int B = 256;
    auto nb = [](long n, int b) { return (int)((n + b - 1) / b); };

    // weight / input prep
    pack_input_kernel<<<nb((long)NNODES * K1PAD, B), B, 0, stream>>>(node_feat, inpad);
    prep_w1_kernel<<<nb(HID * K1PAD, B), B, 0, stream>>>(W1, w1bf);
    {
        int total = (N_LAYERS - 1) * HID * HID;
        prep_w_kernel<<<nb(total, B), B, 0, stream>>>(W, wbf, total);
    }

    const int n_mtiles = (NNODES + 15) / 16;                    // 6250
    dim3 ggrid((n_mtiles + 7) / 8, HID / (16 * NT_PER_WAVE));   // (782, 3)

    for (int layer = 0; layer < N_LAYERS; ++layer) {
        const unsigned short* A  = (layer == 0) ? inpad : hbf;
        const unsigned short* Bm = (layer == 0) ? w1bf : (wbf + (size_t)(layer - 1) * HID * HID);
        const int K              = (layer == 0) ? K1PAD : HID;
        const float* al          = (layer == 0) ? a1 : (a + (size_t)(layer - 1) * N_HEADS * 2 * H_FEAT);

        gemm_bf16_kernel<<<ggrid, 256, 0, stream>>>(A, Bm, z, NNODES, K);
        es_ed_kernel<<<nb((long)NNODES * N_HEADS, B), B, 0, stream>>>(z, al, es, ed);
        init_m_den_kernel<<<nb((long)NNODES * N_HEADS, B), B, 0, stream>>>(menc, den);
        edge_logits_kernel<<<nb((long)NEDGES * N_HEADS, B), B, 0, stream>>>(src, dst, es, ed, ebuf, menc);
        finalize_m_kernel<<<nb((long)NNODES * N_HEADS, B), B, 0, stream>>>(menc);
        hipMemsetAsync(hn, 0, (size_t)NNODES * HID * 4, stream);
        edge_exp_kernel<<<nb((long)NEDGES * N_HEADS, B), B, 0, stream>>>(dst, ebuf, (const float*)menc, exbuf, den);
        edge_aggregate_kernel<<<NEDGES, HID, 0, stream>>>(src, dst, exbuf, den, z, hn);
        elu_convert_kernel<<<nb((long)NNODES * HID, B), B, 0, stream>>>(hn, hf, hbf);
    }

    final_fc_kernel<<<nb((long)NNODES * NUM_CLASSES, B), B, 0, stream>>>(hf, fcW, fcb, (float*)d_out);
}
